// NMT_30374008717734
// MI455X (gfx1250) — compile-verified, matrix operand order checked by reading, not run on hardware
//
#include <hip/hip_runtime.h>
#include <math.h>

// ---------------------------------------------------------------------------
// RNNsearch NMT forward for MI455X (gfx1250, wave32, WMMA).
// - All weights converted once per launch to bf16 [N][K] (54 MB -> resident in
//   the 192 MB L2, so the 150 sequential GRU steps stream weights from L2).
// - Every GEMM runs through v_wmma_f32_16x16x32_bf16 (8x FLOP/instr vs f32
//   WMMA), f32 accumulate, f32 biases/nonlinearities/hidden state.
// - Big time-parallel GEMMs (M=3200): 16x64 tile/wave (4 acc, A reuse).
//   Latency-critical recurrent GEMMs (M=64): 16x16 tile/wave -> 4x more waves
//   in flight and a 4x shorter dependent-WMMA chain per wave.
// ---------------------------------------------------------------------------

#define LSEQ 50
#define BATCH 64
#define EH 1024
#define DH 1024
#define SWE 512
#define AL 1024
#define OUTD 512
#define ROWS (LSEQ * BATCH)          // 3200

typedef __attribute__((ext_vector_type(16))) __bf16 v16bf;
typedef __attribute__((ext_vector_type(8)))  float  v8f;
typedef __attribute__((ext_vector_type(4)))  unsigned int u32x4;

union Frag { v16bf v; u32x4 u[2]; };

__device__ __forceinline__ unsigned short f2bf(float f) {
    union { float f; unsigned int u; } c; c.f = f;
    unsigned int u = c.u;
    return (unsigned short)((u + 0x7FFFu + ((u >> 16) & 1u)) >> 16);
}
__device__ __forceinline__ float sigf(float x) { return 1.0f / (1.0f + expf(-x)); }
__device__ __forceinline__ u32x4 ld128(const unsigned short* p) {
    return *(const u32x4*)p;
}

// ---------------------------------------------------------------------------
// Generic bf16 GEMM:  C[M,N] = A[M,K] * Wt[N,K]^T (+bias) (opt tanh)
// A row-major with row stride lda (elements), Wt row-major [N][K].
// Block = 128 threads (4 waves); each wave owns a 16 x (16*NT) tile.
// grid = (N/(64*NT), M/16).
// Fragment layout per CDNA5 ISA 7.12.2 (16-bit A: lanes 0-15 = rows,
// K split 8*(lane/16); B symmetric with N on lanes).
// ---------------------------------------------------------------------------
template <int NT>
__global__ __launch_bounds__(128) void gemm_bf16_wmma(
    const unsigned short* __restrict__ A, int lda,
    const unsigned short* __restrict__ Wt,
    const float* __restrict__ bias,
    float* __restrict__ C, int ldc,
    int K, int flags)                     // flags bit0: tanh epilogue
{
    const int lane = threadIdx.x & 31;
    const int wave = threadIdx.x >> 5;
    const int m0 = blockIdx.y * 16;
    const int n0 = (blockIdx.x * 4 + wave) * (16 * NT);
    const int mr = lane & 15;
    const int kh = lane >> 4;

    v8f acc[NT] = {};
    const unsigned short* pa = A + (size_t)(m0 + mr) * lda + 8 * kh;
    const unsigned short* pb[NT];
#pragma unroll
    for (int j = 0; j < NT; ++j)
        pb[j] = Wt + (size_t)(n0 + 16 * j + mr) * K + 8 * kh;

    for (int k0 = 0; k0 < K; k0 += 32) {
        Frag a;
        a.u[0] = ld128(pa + k0);
        a.u[1] = ld128(pa + k0 + 16);
        __builtin_prefetch(pb[0] + k0 + 512, 0, 1);   // -> global_prefetch_b8
#pragma unroll
        for (int j = 0; j < NT; ++j) {
            Frag b;
            b.u[0] = ld128(pb[j] + k0);
            b.u[1] = ld128(pb[j] + k0 + 16);
            acc[j] = __builtin_amdgcn_wmma_f32_16x16x32_bf16(false, a.v, false, b.v,
                                                             (short)0, acc[j], false, false);
        }
    }

    const int mbase = m0 + 8 * kh;      // C layout: lane, vgpr i -> M = i + 8*(lane/16)
    const int nb = n0 + (lane & 15);
#pragma unroll
    for (int j = 0; j < NT; ++j) {
        const int n = nb + 16 * j;
        const float bv = bias ? bias[n] : 0.0f;
#pragma unroll
        for (int i = 0; i < 8; ++i) {
            float v = acc[j][i] + bv;
            if (flags & 1) v = tanhf(v);
            C[(size_t)(mbase + i) * ldc + n] = v;
        }
    }
}

// ---------------------------------------------------------------------------
// Weight prep: W[k][n] f32  ->  dst[(rowOff+n)*dstLd + colOff + k] bf16
// ---------------------------------------------------------------------------
__global__ void prep_weight(const float* __restrict__ W, unsigned short* __restrict__ dst,
                            int K, int N, int dstLd, int rowOff, int colOff)
{
    int i = blockIdx.x * blockDim.x + threadIdx.x;
    if (i >= K * N) return;
    int k = i / N, n = i % N;
    dst[(size_t)(rowOff + n) * dstLd + colOff + k] = f2bf(W[i]);
}

__global__ void add_bias3(const float* __restrict__ a, const float* __restrict__ b,
                          const float* __restrict__ c, float* __restrict__ dst, int n)
{
    int i = blockIdx.x * blockDim.x + threadIdx.x;
    if (i < n) dst[i] = a[i] + (b ? b[i] : 0.0f) + (c ? c[i] : 0.0f);
}

__global__ void gather_bf16(const int* __restrict__ idx, const float* __restrict__ emb,
                            unsigned short* __restrict__ dst, int dstLd, int rows, int E)
{
    int i = blockIdx.x * blockDim.x + threadIdx.x;
    if (i >= rows * E) return;
    int r = i / E, e = i % E;
    dst[(size_t)r * dstLd + e] = f2bf(emb[(size_t)idx[r] * E + e]);
}

__global__ void fill_h_zero(float* __restrict__ h, unsigned short* __restrict__ hb, int n)
{
    int i = blockIdx.x * blockDim.x + threadIdx.x;
    if (i < n) { h[i] = 0.0f; hb[i] = 0; }
}

__global__ void f32_to_bf16_strided(const float* __restrict__ src,
                                    unsigned short* __restrict__ dst,
                                    int cols, int dstLd, int n)
{
    int i = blockIdx.x * blockDim.x + threadIdx.x;
    if (i >= n) return;
    int r = i / cols, c = i % cols;
    dst[(size_t)r * dstLd + c] = f2bf(src[i]);
}

// ----- encoder GRU elementwise pieces --------------------------------------
__global__ void enc_combine1(const float* __restrict__ zr, const float* __restrict__ xpre,
                             const float* __restrict__ h, float* __restrict__ zst,
                             unsigned short* __restrict__ rh_bf, int t)
{
    int i = blockIdx.x * blockDim.x + threadIdx.x;
    if (i >= BATCH * EH) return;
    int b = i >> 10, n = i & 1023;
    const float* xr = xpre + (size_t)(t * BATCH + b) * 3072;
    float z = sigf(zr[b * 2048 + n]        + xr[n]);
    float r = sigf(zr[b * 2048 + 1024 + n] + xr[1024 + n]);
    zst[i] = z;
    rh_bf[i] = f2bf(r * h[i]);
}

__global__ void enc_combine2(const float* __restrict__ hpre, const float* __restrict__ xpre,
                             const float* __restrict__ zst, const float* __restrict__ mask,
                             float* __restrict__ h, unsigned short* __restrict__ h_bf,
                             float* __restrict__ xsh_f, unsigned short* __restrict__ xsh_bf,
                             int t, int coloff)
{
    int i = blockIdx.x * blockDim.x + threadIdx.x;
    if (i >= BATCH * EH) return;
    int b = i >> 10, n = i & 1023;
    float hi = hpre[i] + xpre[(size_t)(t * BATCH + b) * 3072 + 2048 + n];
    float z = zst[i];
    float hn = (1.0f - z) * h[i] + z * tanhf(hi);
    float m = mask[t * BATCH + b];
    hn = m * hn + (1.0f - m) * h[i];
    h[i] = hn;
    h_bf[i] = f2bf(hn);
    size_t o = (size_t)(t * BATCH + b) * 2048 + coloff + n;
    xsh_f[o] = hn;
    xsh_bf[o] = f2bf(hn);
}

// ----- attention: score GEMV + softmax + context, fused --------------------
__global__ __launch_bounds__(256) void attention_kernel(
    const float* __restrict__ sa_s, const float* __restrict__ uh,
    const float* __restrict__ a1w, const float* __restrict__ a1b,
    const float* __restrict__ srcs_m, const float* __restrict__ xs_h,
    unsigned short* __restrict__ dec_in_zr, unsigned short* __restrict__ dec_in_h,
    unsigned short* __restrict__ concatC, int t)
{
    __shared__ float sc[LSEQ];
    __shared__ float red[256];
    const int b = blockIdx.x;
    const int tid = threadIdx.x;

    for (int l = 0; l < LSEQ; ++l) {
        float p = 0.0f;
        const float* u = uh + (size_t)(l * BATCH + b) * AL;
        for (int j = tid; j < AL; j += 256)
            p += tanhf(sa_s[b * AL + j] + u[j]) * a1w[j];
        red[tid] = p; __syncthreads();
        for (int s = 128; s > 0; s >>= 1) { if (tid < s) red[tid] += red[tid + s]; __syncthreads(); }
        if (tid == 0) sc[l] = red[0] + a1b[0];
        __syncthreads();
    }
    if (tid == 0) {
        float mx = -1e30f;
        for (int l = 0; l < LSEQ; ++l) mx = fmaxf(mx, sc[l]);
        float s = 0.0f;
        for (int l = 0; l < LSEQ; ++l) { float e = expf(sc[l] - mx) * srcs_m[l * BATCH + b]; sc[l] = e; s += e; }
        float inv = 1.0f / s;
        for (int l = 0; l < LSEQ; ++l) sc[l] *= inv;
    }
    __syncthreads();
    for (int j = tid; j < 2 * EH; j += 256) {
        float acc = 0.0f;
        for (int l = 0; l < LSEQ; ++l)
            acc += sc[l] * xs_h[(size_t)(l * BATCH + b) * 2048 + j];
        unsigned short bv = f2bf(acc);
        dec_in_zr[b * 3072 + 1024 + j] = bv;
        dec_in_h [b * 3072 + 1024 + j] = bv;
        concatC[(size_t)(t * BATCH + b) * 3584 + 1536 + j] = bv;
    }
}

// ----- decoder GRU elementwise pieces --------------------------------------
__global__ void dec_combine1(const float* __restrict__ zr, const float* __restrict__ xpre,
                             const float* __restrict__ s, float* __restrict__ zst,
                             unsigned short* __restrict__ dec_in_h, int t)
{
    int i = blockIdx.x * blockDim.x + threadIdx.x;
    if (i >= BATCH * DH) return;
    int b = i >> 10, n = i & 1023;
    const float* xr = xpre + (size_t)(t * BATCH + b) * 3072;
    float z = sigf(zr[b * 2048 + n]        + xr[n]);
    float r = sigf(zr[b * 2048 + 1024 + n] + xr[1024 + n]);
    zst[i] = z;
    dec_in_h[b * 3072 + n] = f2bf(r * s[i]);
}

__global__ void dec_combine2(const float* __restrict__ hpre, const float* __restrict__ xpre,
                             const float* __restrict__ zst, const float* __restrict__ mask,
                             float* __restrict__ s, unsigned short* __restrict__ dec_in_zr,
                             unsigned short* __restrict__ concatC, int t)
{
    int i = blockIdx.x * blockDim.x + threadIdx.x;
    if (i >= BATCH * DH) return;
    int b = i >> 10, n = i & 1023;
    float hi = hpre[i] + xpre[(size_t)(t * BATCH + b) * 3072 + 2048 + n];
    float z = zst[i];
    float sn = (1.0f - z) * s[i] + z * tanhf(hi);
    float m = mask[t * BATCH + b];
    sn = m * sn + (1.0f - m) * s[i];
    s[i] = sn;
    unsigned short bv = f2bf(sn);
    dec_in_zr[b * 3072 + n] = bv;
    concatC[(size_t)(t * BATCH + b) * 3584 + n] = bv;
}

__global__ void maxout_kernel(const float* __restrict__ lp, const float* __restrict__ trgs_m,
                              float* __restrict__ out)
{
    int i = blockIdx.x * blockDim.x + threadIdx.x;
    if (i >= ROWS * OUTD) return;
    int row = i / OUTD, o = i % OUTD;
    float v = fmaxf(lp[(size_t)row * 1024 + 2 * o], lp[(size_t)row * 1024 + 2 * o + 1]);
    out[i] = v * trgs_m[row];
}

// ---------------------------------------------------------------------------
// Host: input index mapping. Top-level dict insertion order, then `params`
// jax-tree-flattened (sorted keys; each linear dict flattens as {b, w}).
// ---------------------------------------------------------------------------
enum {
    IN_SRCS = 0, IN_TRGS, IN_SRCS_M, IN_TRGS_M,
    A1_B, A1_W,
    BACK_HH_B, BACK_HH_W, BACK_HR_B, BACK_HR_W, BACK_HZ_B, BACK_HZ_W,
    BACK_XH_B, BACK_XH_W, BACK_XR_B, BACK_XR_W, BACK_XZ_B, BACK_XZ_W,
    DEC_CH_B, DEC_CH_W, DEC_CR_B, DEC_CR_W, DEC_CZ_B, DEC_CZ_W,
    DEC_HH_B, DEC_HH_W, DEC_HR_B, DEC_HR_W, DEC_HZ_B, DEC_HZ_W,
    DEC_XH_B, DEC_XH_W, DEC_XR_B, DEC_XR_W, DEC_XZ_B, DEC_XZ_W,
    FORW_HH_B, FORW_HH_W, FORW_HR_B, FORW_HR_W, FORW_HZ_B, FORW_HZ_W,
    FORW_XH_B, FORW_XH_W, FORW_XR_B, FORW_XR_W, FORW_XZ_B, FORW_XZ_W,
    HA_B, HA_W, LC_B, LC_W, LS_B, LS_W, LY_B, LY_W,
    SINIT_B, SINIT_W, SA_B, SA_W, SRC_EMB, TRG_EMB
};

extern "C" void kernel_launch(void* const* d_in, const int* in_sizes, int n_in,
                              void* d_out, int out_size, void* d_ws, size_t ws_size,
                              hipStream_t stream)
{
    (void)in_sizes; (void)n_in; (void)out_size; (void)ws_size;
    auto F = [&](int i) { return (const float*)d_in[i]; };
    auto I = [&](int i) { return (const int*)d_in[i]; };

    char* base = (char*)d_ws; size_t off = 0;
    auto alloc = [&](size_t bytes) -> void* {
        void* r = base + off; off += (bytes + 255) & ~(size_t)255; return r;
    };

    // bf16 weight buffers [N][K]
    unsigned short* Wf_x  = (unsigned short*)alloc((size_t)3072 * 512 * 2);
    unsigned short* Wb_x  = (unsigned short*)alloc((size_t)3072 * 512 * 2);
    unsigned short* Wd_x  = (unsigned short*)alloc((size_t)3072 * 512 * 2);
    unsigned short* Wf_zr = (unsigned short*)alloc((size_t)2048 * 1024 * 2);
    unsigned short* Wb_zr = (unsigned short*)alloc((size_t)2048 * 1024 * 2);
    unsigned short* Wf_hh = (unsigned short*)alloc((size_t)1024 * 1024 * 2);
    unsigned short* Wb_hh = (unsigned short*)alloc((size_t)1024 * 1024 * 2);
    unsigned short* Wd_zr = (unsigned short*)alloc((size_t)2048 * 3072 * 2);
    unsigned short* Wd_h  = (unsigned short*)alloc((size_t)1024 * 3072 * 2);
    unsigned short* Wsa   = (unsigned short*)alloc((size_t)1024 * 1024 * 2);
    unsigned short* Wsini = (unsigned short*)alloc((size_t)1024 * 1024 * 2);
    unsigned short* Wha   = (unsigned short*)alloc((size_t)1024 * 2048 * 2);
    unsigned short* Wout  = (unsigned short*)alloc((size_t)1024 * 3584 * 2);
    // folded biases
    float* bf_x = (float*)alloc(3072 * 4);
    float* bb_x = (float*)alloc(3072 * 4);
    float* bd_x = (float*)alloc(3072 * 4);
    float* bout = (float*)alloc(1024 * 4);
    // activations
    unsigned short* xs_e_bf  = (unsigned short*)alloc((size_t)ROWS * 512 * 2);
    unsigned short* xs_h_bf  = (unsigned short*)alloc((size_t)ROWS * 2048 * 2);
    unsigned short* concat   = (unsigned short*)alloc((size_t)ROWS * 3584 * 2); // [S|ys|C]
    unsigned short* h_bf     = (unsigned short*)alloc((size_t)BATCH * 1024 * 2);
    unsigned short* rh_bf    = (unsigned short*)alloc((size_t)BATCH * 1024 * 2);
    unsigned short* din_zr   = (unsigned short*)alloc((size_t)BATCH * 3072 * 2); // [s|attend]
    unsigned short* din_h    = (unsigned short*)alloc((size_t)BATCH * 3072 * 2); // [r*s|attend]
    float* xpre_f  = (float*)alloc((size_t)ROWS * 3072 * 4);
    float* xpre_b  = (float*)alloc((size_t)ROWS * 3072 * 4);
    float* xpre_d  = (float*)alloc((size_t)ROWS * 3072 * 4);
    float* xs_h_f  = (float*)alloc((size_t)ROWS * 2048 * 4);
    float* uh      = (float*)alloc((size_t)ROWS * 1024 * 4);
    float* h_f32   = (float*)alloc((size_t)BATCH * 1024 * 4);
    float* s_f32   = (float*)alloc((size_t)BATCH * 1024 * 4);
    float* zr_pre  = (float*)alloc((size_t)BATCH * 2048 * 4);
    float* hh_pre  = (float*)alloc((size_t)BATCH * 1024 * 4);
    float* z_st    = (float*)alloc((size_t)BATCH * 1024 * 4);
    float* sa_s    = (float*)alloc((size_t)BATCH * 1024 * 4);
    float* logit   = (float*)alloc((size_t)ROWS * 1024 * 4);

    auto eg = [](size_t n) { return dim3((unsigned)((n + 255) / 256)); };
    auto prep = [&](int w, int K, int N, unsigned short* dst, int ld, int ro, int co) {
        prep_weight<<<eg((size_t)K * N), 256, 0, stream>>>(F(w), dst, K, N, ld, ro, co);
    };
    // Big time-parallel GEMMs: 16x64 tile/wave (compute density).
    auto gemm_big = [&](const unsigned short* A, int lda, const unsigned short* Wt,
                        const float* bias, float* C, int ldc, int M, int N, int K, int flags) {
        dim3 g(N / 256, M / 16);
        gemm_bf16_wmma<4><<<g, dim3(128), 0, stream>>>(A, lda, Wt, bias, C, ldc, K, flags);
    };
    // Latency-critical recurrent GEMMs (M=64): 16x16 tile/wave (max spread).
    auto gemm_rec = [&](const unsigned short* A, int lda, const unsigned short* Wt,
                        const float* bias, float* C, int ldc, int M, int N, int K, int flags) {
        dim3 g(N / 64, M / 16);
        gemm_bf16_wmma<1><<<g, dim3(128), 0, stream>>>(A, lda, Wt, bias, C, ldc, K, flags);
    };

    // ---- weight prep (per-launch, deterministic) ----
    prep(FORW_XZ_W, 512, 1024, Wf_x, 512, 0, 0);
    prep(FORW_XR_W, 512, 1024, Wf_x, 512, 1024, 0);
    prep(FORW_XH_W, 512, 1024, Wf_x, 512, 2048, 0);
    prep(BACK_XZ_W, 512, 1024, Wb_x, 512, 0, 0);
    prep(BACK_XR_W, 512, 1024, Wb_x, 512, 1024, 0);
    prep(BACK_XH_W, 512, 1024, Wb_x, 512, 2048, 0);
    prep(DEC_XZ_W, 512, 1024, Wd_x, 512, 0, 0);
    prep(DEC_XR_W, 512, 1024, Wd_x, 512, 1024, 0);
    prep(DEC_XH_W, 512, 1024, Wd_x, 512, 2048, 0);
    prep(FORW_HZ_W, 1024, 1024, Wf_zr, 1024, 0, 0);
    prep(FORW_HR_W, 1024, 1024, Wf_zr, 1024, 1024, 0);
    prep(FORW_HH_W, 1024, 1024, Wf_hh, 1024, 0, 0);
    prep(BACK_HZ_W, 1024, 1024, Wb_zr, 1024, 0, 0);
    prep(BACK_HR_W, 1024, 1024, Wb_zr, 1024, 1024, 0);
    prep(BACK_HH_W, 1024, 1024, Wb_hh, 1024, 0, 0);
    prep(DEC_HZ_W, 1024, 1024, Wd_zr, 3072, 0, 0);
    prep(DEC_CZ_W, 2048, 1024, Wd_zr, 3072, 0, 1024);
    prep(DEC_HR_W, 1024, 1024, Wd_zr, 3072, 1024, 0);
    prep(DEC_CR_W, 2048, 1024, Wd_zr, 3072, 1024, 1024);
    prep(DEC_HH_W, 1024, 1024, Wd_h, 3072, 0, 0);
    prep(DEC_CH_W, 2048, 1024, Wd_h, 3072, 0, 1024);
    prep(SA_W,    1024, 1024, Wsa,   1024, 0, 0);
    prep(SINIT_W, 1024, 1024, Wsini, 1024, 0, 0);
    prep(HA_W,    2048, 1024, Wha,   2048, 0, 0);
    prep(LS_W, 1024, 1024, Wout, 3584, 0, 0);
    prep(LY_W,  512, 1024, Wout, 3584, 0, 1024);
    prep(LC_W, 2048, 1024, Wout, 3584, 0, 1536);

    // ---- folded biases ----
    add_bias3<<<eg(1024), 256, 0, stream>>>(F(FORW_XZ_B), F(FORW_HZ_B), nullptr, bf_x, 1024);
    add_bias3<<<eg(1024), 256, 0, stream>>>(F(FORW_XR_B), F(FORW_HR_B), nullptr, bf_x + 1024, 1024);
    add_bias3<<<eg(1024), 256, 0, stream>>>(F(FORW_XH_B), F(FORW_HH_B), nullptr, bf_x + 2048, 1024);
    add_bias3<<<eg(1024), 256, 0, stream>>>(F(BACK_XZ_B), F(BACK_HZ_B), nullptr, bb_x, 1024);
    add_bias3<<<eg(1024), 256, 0, stream>>>(F(BACK_XR_B), F(BACK_HR_B), nullptr, bb_x + 1024, 1024);
    add_bias3<<<eg(1024), 256, 0, stream>>>(F(BACK_XH_B), F(BACK_HH_B), nullptr, bb_x + 2048, 1024);
    add_bias3<<<eg(1024), 256, 0, stream>>>(F(DEC_XZ_B), F(DEC_HZ_B), F(DEC_CZ_B), bd_x, 1024);
    add_bias3<<<eg(1024), 256, 0, stream>>>(F(DEC_XR_B), F(DEC_HR_B), F(DEC_CR_B), bd_x + 1024, 1024);
    add_bias3<<<eg(1024), 256, 0, stream>>>(F(DEC_XH_B), F(DEC_HH_B), F(DEC_CH_B), bd_x + 2048, 1024);
    add_bias3<<<eg(1024), 256, 0, stream>>>(F(LS_B), F(LY_B), F(LC_B), bout, 1024);

    // ---- embedding gathers (ys straight into the fused output-concat) ----
    gather_bf16<<<eg((size_t)ROWS * 512), 256, 0, stream>>>(I(IN_SRCS), F(SRC_EMB), xs_e_bf, 512, ROWS, 512);
    gather_bf16<<<eg((size_t)ROWS * 512), 256, 0, stream>>>(I(IN_TRGS), F(TRG_EMB), concat + 1024, 3584, ROWS, 512);

    // ---- time-parallel x-gate preactivations (biases folded) ----
    gemm_big(xs_e_bf, 512, Wf_x, bf_x, xpre_f, 3072, ROWS, 3072, 512, 0);
    gemm_big(xs_e_bf, 512, Wb_x, bb_x, xpre_b, 3072, ROWS, 3072, 512, 0);
    gemm_big(concat + 1024, 3584, Wd_x, bd_x, xpre_d, 3072, ROWS, 3072, 512, 0);

    // ---- forward encoder scan ----
    fill_h_zero<<<eg(BATCH * 1024), 256, 0, stream>>>(h_f32, h_bf, BATCH * 1024);
    for (int t = 0; t < LSEQ; ++t) {
        gemm_rec(h_bf, 1024, Wf_zr, nullptr, zr_pre, 2048, BATCH, 2048, 1024, 0);
        enc_combine1<<<eg(BATCH * 1024), 256, 0, stream>>>(zr_pre, xpre_f, h_f32, z_st, rh_bf, t);
        gemm_rec(rh_bf, 1024, Wf_hh, nullptr, hh_pre, 1024, BATCH, 1024, 1024, 0);
        enc_combine2<<<eg(BATCH * 1024), 256, 0, stream>>>(hh_pre, xpre_f, z_st, F(IN_SRCS_M),
                                                           h_f32, h_bf, xs_h_f, xs_h_bf, t, 0);
    }
    // ---- backward encoder scan ----
    fill_h_zero<<<eg(BATCH * 1024), 256, 0, stream>>>(h_f32, h_bf, BATCH * 1024);
    for (int t = LSEQ - 1; t >= 0; --t) {
        gemm_rec(h_bf, 1024, Wb_zr, nullptr, zr_pre, 2048, BATCH, 2048, 1024, 0);
        enc_combine1<<<eg(BATCH * 1024), 256, 0, stream>>>(zr_pre, xpre_b, h_f32, z_st, rh_bf, t);
        gemm_rec(rh_bf, 1024, Wb_hh, nullptr, hh_pre, 1024, BATCH, 1024, 1024, 0);
        enc_combine2<<<eg(BATCH * 1024), 256, 0, stream>>>(hh_pre, xpre_b, z_st, F(IN_SRCS_M),
                                                           h_f32, h_bf, xs_h_f, xs_h_bf, t, 1024);
    }

    // ---- s0 = tanh(s_init(h0_left)); h_bf currently holds rev_left[0] ----
    gemm_rec(h_bf, 1024, Wsini, F(SINIT_B), s_f32, 1024, BATCH, 1024, 1024, /*tanh*/1);
    f32_to_bf16_strided<<<eg(BATCH * 1024), 256, 0, stream>>>(s_f32, din_zr, 1024, 3072, BATCH * 1024);

    // ---- uh = ha(xs_h) ----
    gemm_big(xs_h_bf, 2048, Wha, F(HA_B), uh, 1024, ROWS, 1024, 2048, 0);

    // ---- decoder scan ----
    for (int t = 0; t < LSEQ; ++t) {
        gemm_rec(din_zr, 3072, Wsa, F(SA_B), sa_s, 1024, BATCH, 1024, 1024, 0);
        attention_kernel<<<dim3(BATCH), dim3(256), 0, stream>>>(
            sa_s, uh, F(A1_W), F(A1_B), F(IN_SRCS_M), xs_h_f, din_zr, din_h, concat, t);
        gemm_rec(din_zr, 3072, Wd_zr, nullptr, zr_pre, 2048, BATCH, 2048, 3072, 0);
        dec_combine1<<<eg(BATCH * 1024), 256, 0, stream>>>(zr_pre, xpre_d, s_f32, z_st, din_h, t);
        gemm_rec(din_h, 3072, Wd_h, nullptr, hh_pre, 1024, BATCH, 1024, 3072, 0);
        dec_combine2<<<eg(BATCH * 1024), 256, 0, stream>>>(hh_pre, xpre_d, z_st, F(IN_TRGS_M),
                                                           s_f32, din_zr, concat, t);
    }

    // ---- fused readout: logit = [S|ys|C] @ [ls|ly|lc]^T + (ls.b+ly.b+lc.b) ----
    gemm_big(concat, 3584, Wout, bout, logit, 1024, ROWS, 1024, 3584, 0);
    maxout_kernel<<<eg((size_t)ROWS * OUTD), 256, 0, stream>>>(logit, F(IN_TRGS_M), (float*)d_out);
}
